// AgnisV5_5497558138957
// MI455X (gfx1250) — matrix operand order repaired and checked by reading, not compiled, not run on hardware
//
#include <hip/hip_runtime.h>
#include <math.h>

// ---------------------------------------------------------------------------
// Model dims (match reference)
// ---------------------------------------------------------------------------
#define D_    768
#define HID_  3072
#define FFN_  2048
#define NH_   8
#define NL_   2
#define T_    64
#define V_    50257
#define B_    16
#define HD_   96
#define ALPHA_ 0.4f

typedef __attribute__((ext_vector_type(16))) _Float16 v16h;
typedef __attribute__((ext_vector_type(8)))  float    v8f;

__device__ __forceinline__ float gelu_exact(float x) {
    return 0.5f * x * (1.0f + erff(x * 0.7071067811865476f));
}
__device__ __forceinline__ float sigmoidf_(float x) {
    return 1.0f / (1.0f + expf(-x));
}

// ---------------------------------------------------------------------------
// WMMA GEMM:  C[M,N] = epilogue( A[M,K] @ B + bias, res )
//   transB==1 : B = W[N,K]^T   (y = x @ W.T, PyTorch-style weight)
//   transB==0 : B = W[K,N]     (y = x @ W)
// epilogue: 0 none | 1 bias+gelu | 2 bias | 3 bias+sigmoid | 5 res+acc+bias
//
// Block = 128 threads = 4 waves. Block tile 32x256; each wave owns a 32x64
// slice: 2 A fragments x 4 B fragments = 8 v_wmma_f32_16x16x32_f16 per K-step.
//
// LDS layouts are chosen so every fragment gather is a 32-byte vector load
// (2 x ds_load_b128):
//   Af[mt][lane][j]  : A staged directly in WMMA fragment order
//   Bs[n][k] (f16)   : a lane's B fragment (col n, K=16h..16h+15) contiguous
//
// Edge handling is CLAMP-based, not predicated: out-of-range A rows / B cols
// stage garbage from a clamped (in-bounds) address; garbage only reaches
// accumulator rows / output columns whose stores are masked (m<M, nn<N).
// This keeps staging branch-free (v_cndmask on the address, no exec churn)
// and EXEC all-ones at every WMMA. K % 32 == 0.
// ---------------------------------------------------------------------------
__global__ __launch_bounds__(128)
void gemm_wmma_f16(const float* __restrict__ A, const float* __restrict__ W,
                   const float* __restrict__ bias, const float* __restrict__ res,
                   float* __restrict__ C, int M, int N, int K,
                   int transB, int ep)
{
    __shared__ _Float16 Af[2 * 32 * 16];   // fragment-order A   2 KB
    __shared__ _Float16 Bs[256 * 32];      // [n][k]            16 KB

    const int tid  = threadIdx.x;
    const int lane = tid & 31;
    const int wv   = tid >> 5;                 // 0..3
    const int nblk = blockIdx.x * 256;         // block N origin
    const int m0   = blockIdx.y * 32;          // block M origin

    const int half = lane >> 4;                // 0: lanes 0-15, 1: lanes 16-31
    const int mr   = lane & 15;

    v8f acc[2][4];
#pragma unroll
    for (int mt = 0; mt < 2; ++mt)
#pragma unroll
        for (int s = 0; s < 4; ++s)
#pragma unroll
            for (int r = 0; r < 8; ++r) acc[mt][s][r] = 0.0f;

    for (int k0 = 0; k0 < K; k0 += 32) {
        // ---- stage A (32 rows x 32 k) in fragment order: 1024 elems ----
        // Af[mt*512 + ln*16 + j] = A[clamp(m0+mt*16+(ln&15))][k0 + ka(ln,j)]
#pragma unroll
        for (int i = 0; i < 8; ++i) {
            int idx = i * 128 + tid;
            int mt  = idx >> 9;
            int ln  = (idx >> 4) & 31;
            int j   = idx & 15;
            int row = m0 + mt * 16 + (ln & 15);
            row = (row < M) ? row : (M - 1);           // clamp: stores masked later
            int ka  = j + 16 * (ln >> 4);
            Af[idx] = (_Float16)A[(long long)row * K + k0 + ka];
        }
        if (k0 + 32 < K)  // prefetch next A K-tile (global_prefetch_b8)
            __builtin_prefetch(A + (long long)(m0 + mr) * K + k0 + 32 + half * 16, 0, 1);

        // ---- stage B (32 k x 256 n) transposed as Bs[n*32+k]: 8192 elems ----
        if (transB) {
#pragma unroll
            for (int i = 0; i < 64; ++i) {
                int idx = i * 128 + tid;
                int n = idx >> 5, k = idx & 31;   // lane varies k: coalesced W[N,K] rows
                int gn = nblk + n;
                gn = (gn < N) ? gn : (N - 1);          // clamp: stores masked later
                Bs[n * 32 + k] = (_Float16)W[(long long)gn * K + k0 + k];
            }
        } else {
#pragma unroll
            for (int i = 0; i < 64; ++i) {
                int idx = i * 128 + tid;
                int k = idx >> 8, n = idx & 255;  // lane varies n: coalesced W[K,N] rows
                int gn = nblk + n;
                gn = (gn < N) ? gn : (N - 1);          // clamp: stores masked later
                Bs[n * 32 + k] = (_Float16)W[(long long)(k0 + k) * N + gn];
            }
        }
        __syncthreads();

        // ---- fragment gathers: pure 32-byte LDS vector loads ----
        v16h af0 = *(const v16h*)&Af[lane * 16];
        v16h af1 = *(const v16h*)&Af[512 + lane * 16];
#pragma unroll
        for (int s = 0; s < 4; ++s) {
            const int nl = wv * 64 + s * 16 + mr;   // local col within block tile
            v16h bf = *(const v16h*)&Bs[nl * 32 + 16 * half];
            acc[0][s] = __builtin_amdgcn_wmma_f32_16x16x32_f16(false, af0, false, bf,
                                                               (short)0, acc[0][s],
                                                               false, false);
            acc[1][s] = __builtin_amdgcn_wmma_f32_16x16x32_f16(false, af1, false, bf,
                                                               (short)0, acc[1][s],
                                                               false, false);
        }
        __syncthreads();
    }

    // ---- epilogue + store (C/D layout: lanes0-15 rows0-7, lanes16-31 rows8-15)
#pragma unroll
    for (int mt = 0; mt < 2; ++mt) {
#pragma unroll
        for (int s = 0; s < 4; ++s) {
            const int nn = nblk + wv * 64 + s * 16 + mr;
            if (nn < N) {
                const float bv = bias ? bias[nn] : 0.0f;
#pragma unroll
                for (int r = 0; r < 8; ++r) {
                    int m = m0 + mt * 16 + 8 * half + r;
                    if (m < M) {
                        float v = acc[mt][s][r];
                        if (ep == 1)      v = gelu_exact(v + bv);
                        else if (ep == 2) v = v + bv;
                        else if (ep == 3) v = sigmoidf_(v + bv);
                        else if (ep == 5) v = res[(long long)m * N + nn] + v + bv;
                        C[(long long)m * N + nn] = v;
                    }
                }
            }
        }
    }
}

// ---------------------------------------------------------------------------
// Embedding: x[bt,:] = emb[token[bt],:] + pos[t,:]
// ---------------------------------------------------------------------------
__global__ __launch_bounds__(256)
void embed_kernel(const int* __restrict__ tok, const float* __restrict__ emb,
                  const float* __restrict__ pos, float* __restrict__ x)
{
    int bt = blockIdx.x;
    int t  = bt % T_;
    long long id = tok[bt];
    for (int d = threadIdx.x; d < D_; d += 256)
        x[bt * D_ + d] = emb[id * D_ + d] + pos[t * D_ + d];
}

// ---------------------------------------------------------------------------
// LayerNorm over rows of length D (one block per row)
// ---------------------------------------------------------------------------
__global__ __launch_bounds__(256)
void ln_kernel(const float* __restrict__ x, const float* __restrict__ g,
               const float* __restrict__ b, float* __restrict__ out, int D)
{
    __shared__ float s1[256], s2[256];
    const long long row = blockIdx.x;
    const float* xr = x + row * D;
    float sum = 0.f, sq = 0.f;
    for (int d = threadIdx.x; d < D; d += 256) { float v = xr[d]; sum += v; sq += v * v; }
    s1[threadIdx.x] = sum; s2[threadIdx.x] = sq;
    __syncthreads();
    for (int st = 128; st > 0; st >>= 1) {
        if (threadIdx.x < st) { s1[threadIdx.x] += s1[threadIdx.x + st];
                                s2[threadIdx.x] += s2[threadIdx.x + st]; }
        __syncthreads();
    }
    float m   = s1[0] / (float)D;
    float var = s2[0] / (float)D - m * m;
    float inv = rsqrtf(var + 1e-5f);
    for (int d = threadIdx.x; d < D; d += 256)
        out[row * D + d] = (xr[d] - m) * inv * g[d] + b[d];
}

// ---------------------------------------------------------------------------
// L2-normalize rows: out = x / max(||x||, 1e-12)
// ---------------------------------------------------------------------------
__global__ __launch_bounds__(256)
void l2_kernel(const float* __restrict__ x, float* __restrict__ out, int D)
{
    __shared__ float s1[256];
    const long long row = blockIdx.x;
    const float* xr = x + row * D;
    float sq = 0.f;
    for (int d = threadIdx.x; d < D; d += 256) { float v = xr[d]; sq += v * v; }
    s1[threadIdx.x] = sq;
    __syncthreads();
    for (int st = 128; st > 0; st >>= 1) {
        if (threadIdx.x < st) s1[threadIdx.x] += s1[threadIdx.x + st];
        __syncthreads();
    }
    float inv = 1.0f / fmaxf(sqrtf(s1[0]), 1e-12f);
    for (int d = threadIdx.x; d < D; d += 256)
        out[row * D + d] = xr[d] * inv;
}

// ---------------------------------------------------------------------------
// Causal MHA for one (batch, head). qkv: [B,T,3D]; out: [B,T,D] head slice.
// ---------------------------------------------------------------------------
__global__ __launch_bounds__(256)
void attn_kernel(const float* __restrict__ qkv, float* __restrict__ out)
{
    __shared__ float qs[T_][HD_];
    __shared__ float ks[16][HD_];
    __shared__ float vs[16][HD_];
    __shared__ float ss[T_][T_];

    const int b = blockIdx.x >> 3;
    const int h = blockIdx.x & 7;
    const float scale = 0.1020620726159658f;  // 1/sqrt(96)

    for (int i = threadIdx.x; i < T_ * HD_; i += 256) {
        int r = i / HD_, d = i % HD_;
        qs[r][d] = qkv[(b * T_ + r) * (3 * D_) + h * HD_ + d];
    }
    __syncthreads();

    for (int kb = 0; kb < 4; ++kb) {
        for (int i = threadIdx.x; i < 16 * HD_; i += 256) {
            int c = i / HD_, d = i % HD_;
            ks[c][d] = qkv[(b * T_ + kb * 16 + c) * (3 * D_) + D_ + h * HD_ + d];
        }
        __syncthreads();
#pragma unroll
        for (int p = 0; p < 4; ++p) {
            int idx = p * 256 + threadIdx.x;       // 1024 (q,k) pairs
            int r = idx >> 4, c = idx & 15;
            int kk = kb * 16 + c;
            float dot = 0.f;
            for (int d = 0; d < HD_; ++d) dot += qs[r][d] * ks[c][d];
            ss[r][kk] = (kk > r) ? -__builtin_inff() : dot * scale;
        }
        __syncthreads();
    }

    if (threadIdx.x < T_) {                         // softmax, 1 thread per row
        int r = threadIdx.x;
        float mx = -__builtin_inff();
        for (int c = 0; c <= r; ++c) mx = fmaxf(mx, ss[r][c]);
        float sm = 0.f;
        for (int c = 0; c < T_; ++c) {
            float e = (c <= r) ? expf(ss[r][c] - mx) : 0.0f;
            ss[r][c] = e; sm += e;
        }
        float inv = 1.0f / sm;
        for (int c = 0; c < T_; ++c) ss[r][c] *= inv;
    }
    __syncthreads();

    float oacc[24];
#pragma unroll
    for (int j = 0; j < 24; ++j) oacc[j] = 0.f;
    for (int kb = 0; kb < 4; ++kb) {
        for (int i = threadIdx.x; i < 16 * HD_; i += 256) {
            int c = i / HD_, d = i % HD_;
            vs[c][d] = qkv[(b * T_ + kb * 16 + c) * (3 * D_) + 2 * D_ + h * HD_ + d];
        }
        __syncthreads();
#pragma unroll
        for (int j = 0; j < 24; ++j) {
            int oi = threadIdx.x + 256 * j;         // 6144 output elems
            int r = oi / HD_, d = oi % HD_;
            float a = 0.f;
            for (int c = 0; c < 16; ++c) a += ss[r][kb * 16 + c] * vs[c][d];
            oacc[j] += a;
        }
        __syncthreads();
    }
#pragma unroll
    for (int j = 0; j < 24; ++j) {
        int oi = threadIdx.x + 256 * j;
        int r = oi / HD_, d = oi % HD_;
        out[(b * T_ + r) * D_ + h * HD_ + d] = oacc[j];
    }
}

// ---------------------------------------------------------------------------
// Scan glue kernels
// ---------------------------------------------------------------------------
__global__ void zero_kernel(float* p, int n)
{
    int i = blockIdx.x * 256 + threadIdx.x;
    if (i < n) p[i] = 0.f;
}

__global__ void tc_kernel(const float* __restrict__ ctx, const float* __restrict__ traw,
                          float* __restrict__ tc, int t)
{
    int i = blockIdx.x * 256 + threadIdx.x;
    if (i < B_ * D_) {
        int b = i / D_, d = i % D_;
        tc[i] = ctx[(b * T_ + t) * D_ + d] + ALPHA_ * traw[i];
    }
}

__global__ void cat_kernel(const float* __restrict__ ctx, const float* __restrict__ cf,
                           float* __restrict__ cat, int t)
{
    int i = blockIdx.x * 256 + threadIdx.x;
    if (i < B_ * 2 * D_) {
        int b = i / (2 * D_), d = i % (2 * D_);
        cat[i] = (d < D_) ? ctx[(b * T_ + t) * D_ + d] : cf[b * D_ + (d - D_)];
    }
}

// h_t = clip(LN(gate*(cf+a*tf)+(1-gate)*ctx_t), -5, 5); write h_prev & H[b,t]
__global__ __launch_bounds__(256)
void finalize_kernel(const float* __restrict__ gate, const float* __restrict__ cf,
                     const float* __restrict__ tf, const float* __restrict__ ctx,
                     const float* __restrict__ og, const float* __restrict__ ob,
                     float* __restrict__ hprev, float* __restrict__ H, int t)
{
    __shared__ float pre[D_];
    __shared__ float s1[256], s2[256];
    const int b = blockIdx.x;
    float sum = 0.f, sq = 0.f;
    for (int d = threadIdx.x; d < D_; d += 256) {
        float g = gate[b * D_ + d];
        float v = g * (cf[b * D_ + d] + ALPHA_ * tf[b * D_ + d])
                + (1.0f - g) * ctx[(b * T_ + t) * D_ + d];
        pre[d] = v; sum += v; sq += v * v;
    }
    s1[threadIdx.x] = sum; s2[threadIdx.x] = sq;
    __syncthreads();
    for (int st = 128; st > 0; st >>= 1) {
        if (threadIdx.x < st) { s1[threadIdx.x] += s1[threadIdx.x + st];
                                s2[threadIdx.x] += s2[threadIdx.x + st]; }
        __syncthreads();
    }
    float m   = s1[0] / (float)D_;
    float var = s2[0] / (float)D_ - m * m;
    float inv = rsqrtf(var + 1e-5f);
    for (int d = threadIdx.x; d < D_; d += 256) {
        float v = (pre[d] - m) * inv * og[d] + ob[d];
        v = fminf(5.0f, fmaxf(-5.0f, v));
        hprev[b * D_ + d] = v;
        H[(b * T_ + t) * D_ + d] = v;
    }
}

// ---------------------------------------------------------------------------
// Host launcher
// ---------------------------------------------------------------------------
static inline void gemm(hipStream_t s, const float* A, const float* W,
                        const float* bias, const float* res, float* C,
                        int M, int N, int K, int transB, int ep)
{
    dim3 grid((N + 255) / 256, (M + 31) / 32);
    gemm_wmma_f16<<<grid, 128, 0, s>>>(A, W, bias, res, C, M, N, K, transB, ep);
}

extern "C" void kernel_launch(void* const* d_in, const int* in_sizes, int n_in,
                              void* d_out, int out_size, void* d_ws, size_t ws_size,
                              hipStream_t stream)
{
    (void)in_sizes; (void)n_in; (void)out_size; (void)ws_size;

    const int*   tok        = (const int*)  d_in[0];
    const float* emb        = (const float*)d_in[1];
    const float* pos_emb    = (const float*)d_in[2];
    const float* attn_in_w  = (const float*)d_in[3];
    const float* attn_in_b  = (const float*)d_in[4];
    const float* attn_out_w = (const float*)d_in[5];
    const float* attn_out_b = (const float*)d_in[6];
    const float* ff_w1      = (const float*)d_in[7];
    const float* ff_b1      = (const float*)d_in[8];
    const float* ff_w2      = (const float*)d_in[9];
    const float* ff_b2      = (const float*)d_in[10];
    const float* n1_g       = (const float*)d_in[11];
    const float* n1_b       = (const float*)d_in[12];
    const float* n2_g       = (const float*)d_in[13];
    const float* n2_b       = (const float*)d_in[14];
    const float* enc_g      = (const float*)d_in[15];
    const float* enc_b      = (const float*)d_in[16];
    const float* V0         = (const float*)d_in[17];
    const float* b0         = (const float*)d_in[18];
    const float* V1         = (const float*)d_in[19];
    const float* b1         = (const float*)d_in[20];
    const float* cp_w1      = (const float*)d_in[21];
    const float* cp_b1      = (const float*)d_in[22];
    const float* cp_w2      = (const float*)d_in[23];
    const float* cp_b2      = (const float*)d_in[24];
    const float* gate_w     = (const float*)d_in[25];
    const float* gate_b     = (const float*)d_in[26];
    const float* temp_w     = (const float*)d_in[27];
    const float* Rm         = (const float*)d_in[28];
    const float* out_g      = (const float*)d_in[29];
    const float* out_b      = (const float*)d_in[30];
    const float* lm_head    = (const float*)d_in[31];
    float* logits = (float*)d_out;

    // workspace carve-up (floats)
    float* ws = (float*)d_ws;
    size_t o = 0;
    auto alloc = [&](size_t n) { float* p = ws + o; o += n; return p; };
    const int BT = B_ * T_;                  // 1024
    float* x     = alloc((size_t)BT * D_);
    float* xn    = alloc((size_t)BT * D_);
    float* qkv   = alloc((size_t)BT * 3 * D_);
    float* attnb = alloc((size_t)BT * D_);
    float* ffh   = alloc((size_t)BT * FFN_);
    float* ctx   = alloc((size_t)BT * D_);
    float* H     = alloc((size_t)BT * D_);
    float* hprev = alloc((size_t)B_ * D_);
    float* traw  = alloc((size_t)B_ * D_);
    float* tcb   = alloc((size_t)B_ * D_);
    float* hsh   = alloc((size_t)B_ * HID_);
    float* core  = alloc((size_t)B_ * D_);
    float* blend = alloc((size_t)B_ * D_);
    float* tmp1  = alloc((size_t)B_ * D_);
    float* cfb   = alloc((size_t)B_ * D_);
    float* tfb   = alloc((size_t)B_ * D_);
    float* catb  = alloc((size_t)B_ * 2 * D_);
    float* gateb = alloc((size_t)B_ * D_);

    // ---------------- encoder ----------------
    embed_kernel<<<BT, 256, 0, stream>>>(tok, emb, pos_emb, x);
    for (int l = 0; l < NL_; ++l) {
        ln_kernel<<<BT, 256, 0, stream>>>(x, n1_g + l * D_, n1_b + l * D_, xn, D_);
        gemm(stream, xn, attn_in_w + (size_t)l * 3 * D_ * D_, attn_in_b + l * 3 * D_,
             nullptr, qkv, BT, 3 * D_, D_, /*transB=*/1, /*ep=*/2);
        attn_kernel<<<B_ * NH_, 256, 0, stream>>>(qkv, attnb);
        gemm(stream, attnb, attn_out_w + (size_t)l * D_ * D_, attn_out_b + l * D_,
             x, x, BT, D_, D_, 1, /*ep: res+acc+bias*/ 5);
        ln_kernel<<<BT, 256, 0, stream>>>(x, n2_g + l * D_, n2_b + l * D_, xn, D_);
        gemm(stream, xn, ff_w1 + (size_t)l * FFN_ * D_, ff_b1 + l * FFN_,
             nullptr, ffh, BT, FFN_, D_, 1, /*bias+gelu*/ 1);
        gemm(stream, ffh, ff_w2 + (size_t)l * D_ * FFN_, ff_b2 + l * D_,
             x, x, BT, D_, FFN_, 1, 5);
    }
    ln_kernel<<<BT, 256, 0, stream>>>(x, enc_g, enc_b, xn, D_);
    l2_kernel<<<BT, 256, 0, stream>>>(xn, ctx, D_);

    // ---------------- recurrent scan ----------------
    // Forward-pass identity of stop_gradient => core_set == core_sh => blend == core_sh.
    zero_kernel<<<(B_ * D_ + 255) / 256, 256, 0, stream>>>(hprev, B_ * D_);
    for (int t = 0; t < T_; ++t) {
        gemm(stream, hprev, Rm, nullptr, nullptr, traw, B_, D_, D_, /*[K,N]*/ 0, 0);
        tc_kernel<<<(B_ * D_ + 255) / 256, 256, 0, stream>>>(ctx, traw, tcb, t);
        gemm(stream, tcb, V0, b0, nullptr, hsh, B_, HID_, D_, 0, 1);        // gelu(tc@V0+b0)
        gemm(stream, hsh, V1, b1, nullptr, core, B_, D_, HID_, 0, 1);       // gelu(.@V1+b1)
        l2_kernel<<<B_, 256, 0, stream>>>(core, blend, D_);                 // == blend
        gemm(stream, blend, cp_w1, cp_b1, nullptr, tmp1, B_, D_, D_, 1, 1); // gelu(.@cp1.T+b)
        gemm(stream, tmp1, cp_w2, cp_b2, nullptr, cfb, B_, D_, D_, 1, 2);   // core_proj
        gemm(stream, traw, temp_w, nullptr, nullptr, tfb, B_, D_, D_, 1, 0);// temporal_proj
        cat_kernel<<<(B_ * 2 * D_ + 255) / 256, 256, 0, stream>>>(ctx, cfb, catb, t);
        gemm(stream, catb, gate_w, gate_b, nullptr, gateb, B_, D_, 2 * D_, 1, 3); // sigmoid
        finalize_kernel<<<B_, 256, 0, stream>>>(gateb, cfb, tfb, ctx, out_g, out_b,
                                                hprev, H, t);
    }

    // ---------------- logits: [1024,768] @ lm_head.T -> d_out[B,T,V] ----------------
    gemm(stream, H, lm_head, nullptr, nullptr, logits, BT, V_, D_, 1, 0);
}
